// LSTM_45878840656311
// MI455X (gfx1250) — matrix operand order repaired and checked by reading, not compiled, not run on hardware
//
#include <hip/hip_runtime.h>
#include <hip/hip_bf16.h>
#include <cstddef>
#include <cstdint>

// Problem constants (match reference)
#define BB 512
#define TT 1024
#define DD 128
#define HH 512
#define CC 10
#define KT_N 20           // (D+H)/32
#define NT_N 32           // H / 16
#define NBLK 64           // persistent grid: 32 nt * 2 row-groups
#define NWAVE_TOTAL 512   // 64 blocks * 8 waves

typedef __attribute__((ext_vector_type(16))) __bf16 v16bf;
typedef __attribute__((ext_vector_type(8)))  float  v8f;

// ---------------------------------------------------------------------------
// Workspace layout (bytes)
//   h0    : fp32 [512][512]              @ 0          (1 MiB)   (zeroed)
//   c     : fp32 [512][512]              @ 1 MiB      (1 MiB)   (zeroed)
//   bar   : u32 grid barrier             @ 2 MiB                (zeroed)
//   h1    : fp32 [512][512]              @ 2 MiB+64K  (1 MiB)
//   wfrag : bf16 [4][32 nt][20 kt][512]  @ ...        (2.5 MiB)
//   bias  : fp32 [4][512]                @ ...        (8 KiB)
// ---------------------------------------------------------------------------
#define H0_OFF  ((size_t)0)
#define C_OFF   ((size_t)(1u << 20))
#define BAR_OFF ((size_t)(2u << 20))
#define H1_OFF  ((size_t)(BAR_OFF + 65536))
#define WF_OFF  ((size_t)(H1_OFF + (1u << 20)))
#define BI_OFF  ((size_t)(WF_OFF + 2621440))

__device__ __forceinline__ float sigmoid_fast(float x) {
    return 1.0f / (1.0f + __expf(-x));
}

// ---------------------------------------------------------------------------
// Pack gate weights into WMMA B-fragments (bf16, lane-major, 1 KiB/frag).
// B-tile 32(K) x 16(N). Lane l: n = nt*16 + (l&15); 16 consecutive K values
// starting at kt*32 + (l>>4)*16.  One wave per fragment; 2560 fragments.
// ---------------------------------------------------------------------------
__global__ void pack_w_kernel(const float* __restrict__ W_ix, const float* __restrict__ W_ih,
                              const float* __restrict__ W_fx, const float* __restrict__ W_fh,
                              const float* __restrict__ W_ox, const float* __restrict__ W_oh,
                              const float* __restrict__ W_gx, const float* __restrict__ W_gh,
                              __bf16* __restrict__ wfrag) {
    int wid  = (blockIdx.x * blockDim.x + threadIdx.x) >> 5;   // 0..2559
    int lane = threadIdx.x & 31;
    if (wid >= 4 * NT_N * KT_N) return;

    int g  = wid / (NT_N * KT_N);
    int nt = (wid / KT_N) % NT_N;
    int kt = wid % KT_N;

    const float* Wx = (g == 0) ? W_ix : (g == 1) ? W_fx : (g == 2) ? W_ox : W_gx;
    const float* Wh = (g == 0) ? W_ih : (g == 1) ? W_fh : (g == 2) ? W_oh : W_gh;

    int n     = nt * 16 + (lane & 15);
    int kbase = kt * 32 + (lane >> 4) * 16;

    v16bf pack;
#pragma unroll
    for (int j = 0; j < 16; ++j) {
        int k = kbase + j;
        float v = (k < DD) ? Wx[(size_t)k * HH + n]
                           : Wh[(size_t)(k - DD) * HH + n];
        pack[j] = (__bf16)v;
    }
    *(v16bf*)(wfrag + (size_t)wid * 512 + lane * 16) = pack;
}

__global__ void pack_bias_kernel(const float* __restrict__ B_i, const float* __restrict__ B_f,
                                 const float* __restrict__ B_o, const float* __restrict__ B_g,
                                 float* __restrict__ bias) {
    int tid = blockIdx.x * blockDim.x + threadIdx.x;
    if (tid >= 4 * HH) return;
    int g = tid >> 9, n = tid & 511;
    const float* B = (g == 0) ? B_i : (g == 1) ? B_f : (g == 2) ? B_o : B_g;
    bias[tid] = B[n];
}

// ---------------------------------------------------------------------------
// Build one 16x32 bf16 A-fragment in registers from fp32 rows.
// Lane l: m = tile*16 + (l&15); values 0..15: K = 2*koff..2*koff+15
// (koff = (l>>4)*8), contiguous K order matching the packed B fragments.
// ---------------------------------------------------------------------------
__device__ __forceinline__ v16bf make_afrag(const float* base, int koff) {
    v8f lo = *(const v8f*)(base + 2 * koff);
    v8f hi = *(const v8f*)(base + 2 * koff + 8);
    v16bf a;
#pragma unroll
    for (int j = 0; j < 8; ++j) {
        a[j]     = (__bf16)lo[j];
        a[8 + j] = (__bf16)hi[j];
    }
    return a;
}

// ---------------------------------------------------------------------------
// Persistent LSTM: 64 blocks x 256 threads, all 1024 timesteps in one launch.
//  - block b: nt = b>>1 (16 hidden cols), rg = b&1 (256 batch rows)
//  - wave w: rows bt0 = rg*16 + w*2, bt1 = bt0+1  (32 rows, 8 accumulators)
//  - weights for this nt (80 KiB) staged ONCE into LDS via async global->LDS,
//    read every step with ds_load_b128 (L2 weight traffic ~5 MB total)
//  - grid-wide release/acquire barrier between steps; h ping-pongs h0<->h1
// ---------------------------------------------------------------------------
__global__ void __launch_bounds__(256)
lstm_persistent_kernel(const float* __restrict__ x,
                       const __bf16* __restrict__ wfrag,
                       const float*  __restrict__ bias,
                       float* __restrict__ h0, float* __restrict__ h1,
                       float* __restrict__ c,  unsigned* __restrict__ bar) {
    extern __shared__ char smem_raw[];          // 4 gates * 20 kt * 1 KiB = 80 KiB

    const int tid  = threadIdx.x;
    const int lane = tid & 31;
    const int wave = tid >> 5;
    const int nt   = blockIdx.x >> 1;
    const int rg   = blockIdx.x & 1;

    // ---- one-time: stage this nt's weight fragments into LDS (async) ----
    {
        unsigned lds_base = (unsigned)(uintptr_t)(void*)smem_raw;
#pragma unroll
        for (int g = 0; g < 4; ++g) {
            // 20 contiguous KiB per gate in wfrag for fixed (g, nt)
            const char* src = (const char*)wfrag + (size_t)((g * NT_N + nt) * KT_N) * 1024;
            unsigned    dst = lds_base + g * 20480;
            for (int off = tid * 16; off < 20480; off += 256 * 16) {
                const char* ga = src + off;
                unsigned    la = dst + off;
                asm volatile("global_load_async_to_lds_b128 %0, %1, off"
                             :: "v"(la), "v"(ga) : "memory");
            }
        }
        asm volatile("s_wait_asynccnt 0x0" ::: "memory");
        __syncthreads();
    }

    // ---- per-wave geometry ----
    const int bt0  = rg * 16 + wave * 2;
    const int bt1  = bt0 + 1;
    const int mrow = lane & 15;
    const int koff = (lane >> 4) * 8;
    const int mA   = bt0 * 16 + mrow;
    const int mB   = bt1 * 16 + mrow;
    const int n    = nt * 16 + mrow;

    const float* xA = x + (size_t)mA * (TT * DD);
    const float* xB = x + (size_t)mB * (TT * DD);

    const float b_i = bias[n];
    const float b_f = bias[HH + n];
    const float b_o = bias[2 * HH + n];
    const float b_g = bias[3 * HH + n];

    // ---- recurrence ----
    for (int t = 0; t < TT; ++t) {
        const float* hin  = (t & 1) ? h1 : h0;
        float*       hout = (t & 1) ? h0 : h1;
        const float* xtA  = xA + (size_t)t * DD;
        const float* xtB  = xB + (size_t)t * DD;
        const float* hA   = hin + (size_t)mA * HH - DD;   // so col index = kt*32
        const float* hB   = hin + (size_t)mB * HH - DD;

        if (t + 1 < TT) {   // warm L2/WGP$ for next step's x slice
            __builtin_prefetch((const void*)(xtA + DD), 0, 1);
            __builtin_prefetch((const void*)(xtB + DD), 0, 1);
        }

        // accumulators pre-loaded with bias (C-operand trick)
        v8f acc[8];   // [gate*2 + rowpair]
#pragma unroll
        for (int r = 0; r < 8; ++r) {
            acc[0][r] = b_i; acc[1][r] = b_i;
            acc[2][r] = b_f; acc[3][r] = b_f;
            acc[4][r] = b_o; acc[5][r] = b_o;
            acc[6][r] = b_g; acc[7][r] = b_g;
        }

#pragma unroll 4
        for (int kt = 0; kt < KT_N; ++kt) {
            const float* baseA = (kt < 4) ? (xtA + kt * 32) : (hA + kt * 32);
            const float* baseB = (kt < 4) ? (xtB + kt * 32) : (hB + kt * 32);
            v16bf aA = make_afrag(baseA, koff);
            v16bf aB = make_afrag(baseB, koff);
#pragma unroll
            for (int g = 0; g < 4; ++g) {
                v16bf bw = *((const v16bf*)(smem_raw + (size_t)(g * KT_N + kt) * 1024) + lane);
                acc[2 * g]     = __builtin_amdgcn_wmma_f32_16x16x32_bf16(
                    false, aA, false, bw, (short)0, acc[2 * g],     false, false);
                acc[2 * g + 1] = __builtin_amdgcn_wmma_f32_16x16x32_bf16(
                    false, aB, false, bw, (short)0, acc[2 * g + 1], false, false);
            }
        }

        // fused LSTM cell update; each (m,n) owned by exactly this wave
#pragma unroll
        for (int p = 0; p < 2; ++p) {
            const int m0 = (p ? bt1 : bt0) * 16 + koff;   // C/D: M = r + 8*(lane>>4)
#pragma unroll
            for (int r = 0; r < 8; ++r) {
                size_t idx = (size_t)(m0 + r) * HH + n;
                float iv = sigmoid_fast(acc[0 + p][r]);
                float fv = sigmoid_fast(acc[2 + p][r]);
                float ov = sigmoid_fast(acc[4 + p][r]);
                float gv = tanhf(acc[6 + p][r]);
                float cn = fv * c[idx] + iv * gv;
                c[idx]    = cn;
                hout[idx] = ov * tanhf(cn);
            }
        }

        // device-wide split barrier: per-wave release (waits wave STOREcnt),
        // per-wave acquire spin (invalidates L0 before next step's h reads)
        if (lane == 0)
            __hip_atomic_fetch_add(bar, 1u, __ATOMIC_RELEASE, __HIP_MEMORY_SCOPE_AGENT);
        const unsigned target = (unsigned)NWAVE_TOTAL * (unsigned)(t + 1);
        while (__hip_atomic_load(bar, __ATOMIC_ACQUIRE, __HIP_MEMORY_SCOPE_AGENT) < target)
            __builtin_amdgcn_s_sleep(16);
    }
}

// ---------------------------------------------------------------------------
// Head: logits = h @ W_ph + B_y ; softmax over C=10. One wave per batch row.
// ---------------------------------------------------------------------------
__global__ void head_kernel(const float* __restrict__ h, const float* __restrict__ W_ph,
                            const float* __restrict__ B_y, float* __restrict__ out) {
    __shared__ float logits[CC];
    int b    = blockIdx.x;
    int lane = threadIdx.x;
    if (lane < CC) {
        float acc = B_y[lane];
        const float* hr = h + (size_t)b * HH;
        for (int k = 0; k < HH; ++k) acc += hr[k] * W_ph[(size_t)k * CC + lane];
        logits[lane] = acc;
    }
    __syncthreads();
    if (lane < CC) {
        float mx = logits[0];
#pragma unroll
        for (int j = 1; j < CC; ++j) mx = fmaxf(mx, logits[j]);
        float s = 0.f;
#pragma unroll
        for (int j = 0; j < CC; ++j) s += __expf(logits[j] - mx);
        out[(size_t)b * CC + lane] = __expf(logits[lane] - mx) / s;
    }
}

// ---------------------------------------------------------------------------
extern "C" void kernel_launch(void* const* d_in, const int* in_sizes, int n_in,
                              void* d_out, int out_size, void* d_ws, size_t ws_size,
                              hipStream_t stream) {
    const float* x    = (const float*)d_in[0];
    const float* W_ix = (const float*)d_in[1];
    const float* W_ih = (const float*)d_in[2];
    const float* W_fx = (const float*)d_in[3];
    const float* W_fh = (const float*)d_in[4];
    const float* W_ox = (const float*)d_in[5];
    const float* W_oh = (const float*)d_in[6];
    const float* W_gx = (const float*)d_in[7];
    const float* W_gh = (const float*)d_in[8];
    const float* W_ph = (const float*)d_in[9];
    const float* B_i  = (const float*)d_in[10];
    const float* B_f  = (const float*)d_in[11];
    const float* B_o  = (const float*)d_in[12];
    const float* B_g  = (const float*)d_in[13];
    const float* B_y  = (const float*)d_in[14];

    char* ws = (char*)d_ws;
    float*    h0    = (float*)(ws + H0_OFF);
    float*    c_st  = (float*)(ws + C_OFF);
    unsigned* bar   = (unsigned*)(ws + BAR_OFF);
    float*    h1    = (float*)(ws + H1_OFF);
    __bf16*   wfrag = (__bf16*)(ws + WF_OFF);
    float*    bias  = (float*)(ws + BI_OFF);

    // zero h0, c, barrier (contiguous region)
    hipMemsetAsync(ws, 0, BAR_OFF + 4096, stream);

    // one-time packing
    pack_w_kernel<<<(2560 * 32) / 128, 128, 0, stream>>>(W_ix, W_ih, W_fx, W_fh,
                                                         W_ox, W_oh, W_gx, W_gh, wfrag);
    pack_bias_kernel<<<16, 128, 0, stream>>>(B_i, B_f, B_o, B_g, bias);

    // allow 80 KiB dynamic LDS (CDNA5 WGP has 320 KiB)
    hipFuncSetAttribute((const void*)lstm_persistent_kernel,
                        hipFuncAttributeMaxDynamicSharedMemorySize, 81920);

    // entire recurrence in ONE launch
    lstm_persistent_kernel<<<NBLK, 256, 81920, stream>>>(x, wfrag, bias, h0, h1, c_st, bar);

    // classification head + softmax (final h is in h0 after 1024 steps)
    head_kernel<<<BB, 32, 0, stream>>>(h0, W_ph, B_y, (float*)d_out);
}
